// ConvNetLayer_57251914056251
// MI455X (gfx1250) — matrix operand order repaired and checked by reading, not compiled, not run on hardware
//
#include <hip/hip_runtime.h>
#include <stdint.h>

#define NN 16384
#define DD 128
#define BM 64            // i-rows per block  -> 256 blocks
#define KS 128           // j (K) per stage

typedef __attribute__((ext_vector_type(16))) __bf16 v16bf;
typedef __attribute__((ext_vector_type(8)))  float  v8f;

union Frag { uint4 q[2]; v16bf v; };

__device__ __forceinline__ unsigned f32_bf16(float f) {
    union { float f; unsigned u; } c; c.f = f;
    return (c.u + 0x7FFFu + ((c.u >> 16) & 1u)) >> 16;   // round-to-nearest-even
}

// ---------------- pre-pass: x[N][D] f32 -> xT[D][N] bf16 ----------------
__global__ void __launch_bounds__(256) xpose_kernel(const float* __restrict__ x,
                                                    unsigned short* __restrict__ xT) {
    __shared__ unsigned short tile[32][136];
    const int j0 = blockIdx.x * 128;
    const int d0 = blockIdx.y * 32;
    const int t  = threadIdx.x;
    {
        const int d = t & 31, jr = t >> 5;
#pragma unroll
        for (int k = 0; k < 16; ++k) {
            const int j = jr + 8 * k;
            tile[d][j] = (unsigned short)f32_bf16(x[(size_t)(j0 + j) * DD + d0 + d]);
        }
    }
    __syncthreads();
    {
        const int jg = (t & 31) * 4, dr = t >> 5;
#pragma unroll
        for (int k = 0; k < 4; ++k) {
            const int dd = dr + 8 * k;
            uint2 v = *(const uint2*)&tile[dd][jg];
            *(uint2*)&xT[(size_t)(d0 + dd) * NN + j0 + jg] = v;
        }
    }
}

// ---------------- fused GCN layer ----------------
// out[i,e] = relu( sum_d ( (1/deg_i) * sum_j adj[j,i]*x[j,d] ) * U[e,d] )
__global__ void __launch_bounds__(256) gcn_kernel(const int* __restrict__ adj,
                                                  const unsigned short* __restrict__ xT,
                                                  const float* __restrict__ U,
                                                  float* __restrict__ out) {
    __shared__ __align__(16) unsigned char smem[52736];
    unsigned short* adjA = (unsigned short*)smem;            // [64][136]   main A tile
    unsigned short* xB   = (unsigned short*)(smem + 17408);  // [128][136]  main B tile
    unsigned short* aggB = (unsigned short*)smem;            // [64][128]   epilogue A (aliases adjA)
    unsigned short* Ub   = (unsigned short*)(smem + 16384);  // [128][128]  epilogue B (aliases xB)
    float*          sdeg = (float*)(smem + 52224);           // [64]

    const int t    = threadIdx.x;
    const int lane = t & 31;
    const int wave = t >> 5;
    const int m0   = (wave >> 1) * 16;   // 4x2 wave grid: 16(M) x 64(N) per wave
    const int n0   = (wave & 1) * 64;
    const int r16  = lane & 15;
    const int hi   = lane >> 4;
    const int ib   = blockIdx.x * BM;

    // staging assignments
    const int g   = t & 15;              // adjacency: 4 i-columns per thread
    const int jrp = t >> 4;              // 0..15 (j pair slot)
    const int jg8 = (t & 15) * 8;        // x copy: 8 j per thread
    const int dq  = t >> 4;              // 0..15

    const unsigned char* abase = (const unsigned char*)adj;
    unsigned aoff = (unsigned)(((2u * jrp) * NN + ib + 4 * g) * 4u);
    unsigned xoff = (unsigned)((dq * NN + jg8) * 2u);
    const unsigned xb0 = (unsigned)(uintptr_t)(xB + dq * 136 + jg8);  // LDS byte address

    unsigned dpk0 = 0, dpk1 = 0, dpk2 = 0, dpk3 = 0;   // packed degree partials (lo/hi 16-bit)
    v8f acc[4] = {};

    for (int jt = 0; jt < NN; jt += KS) {
        // ---- stage A: adj[j, ib+i] -> adjA[i][j] (bf16 0/1), packed degree count ----
#pragma unroll
        for (int k = 0; k < 4; ++k) {
            const unsigned char* p = abase + (aoff + k * (32u * NN * 4u));
            const int4 a0 = *(const int4*)p;             // row j
            const int4 a1 = *(const int4*)(p + NN * 4);  // row j+1
            const int jb = 2 * jrp + 32 * k;             // even j within stage
            const unsigned p0 = (unsigned)a0.x + ((unsigned)a1.x << 16);
            const unsigned p1 = (unsigned)a0.y + ((unsigned)a1.y << 16);
            const unsigned p2 = (unsigned)a0.z + ((unsigned)a1.z << 16);
            const unsigned p3 = (unsigned)a0.w + ((unsigned)a1.w << 16);
            dpk0 += p0; dpk1 += p1; dpk2 += p2; dpk3 += p3;
            *(unsigned*)(adjA + (4 * g + 0) * 136 + jb) = p0 * 0x3F80u;  // 0/1 -> bf16 pair
            *(unsigned*)(adjA + (4 * g + 1) * 136 + jb) = p1 * 0x3F80u;
            *(unsigned*)(adjA + (4 * g + 2) * 136 + jb) = p2 * 0x3F80u;
            *(unsigned*)(adjA + (4 * g + 3) * 136 + jb) = p3 * 0x3F80u;
        }
        aoff += (unsigned)KS * NN * 4u;
        // ---- stage B: async copy xT[d, jt..jt+127] -> xB[d][j] (ASYNCcnt path) ----
#pragma unroll
        for (int k = 0; k < 8; ++k) {
            const unsigned lds = xb0 + 4352u * (unsigned)k;            // (dq+16k)*272 + jg8*2
            const unsigned vof = xoff + (unsigned)k * (16u * NN * 2u); // global byte offset
            asm volatile("global_load_async_to_lds_b128 %0, %1, %2"
                         :: "v"(lds), "v"(vof), "s"(xT) : "memory");
        }
        xoff += (unsigned)KS * 2u;
        asm volatile("s_wait_asynccnt 0x0" ::: "memory");
        __syncthreads();
        // ---- compute: 16 WMMAs per wave per stage ----
#pragma unroll
        for (int kb = 0; kb < 4; ++kb) {
            Frag af;
            {
                const unsigned char* rp =
                    (const unsigned char*)(adjA + (m0 + r16) * 136) + kb * 64 + hi * 16;
                af.q[0] = *(const uint4*)rp;
                af.q[1] = *(const uint4*)(rp + 32);
            }
            Frag bfr[4];
#pragma unroll
            for (int ni = 0; ni < 4; ++ni) {
                const unsigned char* rp =
                    (const unsigned char*)(xB + (n0 + ni * 16 + r16) * 136) + kb * 64 + hi * 16;
                bfr[ni].q[0] = *(const uint4*)rp;
                bfr[ni].q[1] = *(const uint4*)(rp + 32);
            }
#pragma unroll
            for (int ni = 0; ni < 4; ++ni)
                acc[ni] = __builtin_amdgcn_wmma_f32_16x16x32_bf16(
                    false, af.v, false, bfr[ni].v, (short)0, acc[ni], false, false);
            // group 10 DS reads ahead of the 4 WMMAs -> 4 live B frags, pipelined waits
            __builtin_amdgcn_sched_group_barrier(0x100, 10, 0);
            __builtin_amdgcn_sched_group_barrier(0x008, 4, 0);
        }
        __syncthreads();
    }

    // ---- degree reduction (16 threads contribute per column) ----
    if (t < BM) sdeg[t] = 0.0f;
    __syncthreads();
    atomicAdd(&sdeg[4 * g + 0], (float)((dpk0 & 0xFFFFu) + (dpk0 >> 16)));
    atomicAdd(&sdeg[4 * g + 1], (float)((dpk1 & 0xFFFFu) + (dpk1 >> 16)));
    atomicAdd(&sdeg[4 * g + 2], (float)((dpk2 & 0xFFFFu) + (dpk2 >> 16)));
    atomicAdd(&sdeg[4 * g + 3], (float)((dpk3 & 0xFFFFu) + (dpk3 >> 16)));
    __syncthreads();
    if (t < BM) { const float d = sdeg[t]; sdeg[t] = (d > 0.0f) ? 1.0f / d : 0.0f; }
    __syncthreads();

    // ---- scale by 1/deg, spill agg tile as bf16 (row-major [i][d]) ----
#pragma unroll
    for (int r = 0; r < 8; ++r) {
        const int m = m0 + hi * 8 + r;
        const float inv = sdeg[m];
#pragma unroll
        for (int ni = 0; ni < 4; ++ni) {
            const int c = n0 + ni * 16 + r16;
            aggB[m * 128 + c] = (unsigned short)f32_bf16(acc[ni][r] * inv);
        }
    }
    __syncthreads();

    // ---- load U rows as epilogue-B (column e of B2 == row e of U, K-contiguous) ----
#pragma unroll
    for (int k = 0; k < 32; ++k) {
        const int p = t + 256 * k;     // 8192 f32-pairs
        const int e = p >> 6, d2 = p & 63;
        const float2 uv = *(const float2*)(U + e * 128 + 2 * d2);
        *(unsigned*)(Ub + e * 128 + 2 * d2) = f32_bf16(uv.x) | (f32_bf16(uv.y) << 16);
    }
    __syncthreads();

    // ---- out = relu(agg @ U^T): 16 WMMAs per wave ----
    v8f oacc[4] = {};
#pragma unroll
    for (int kb = 0; kb < 4; ++kb) {
        Frag af;
        {
            const unsigned char* rp =
                (const unsigned char*)(aggB + (m0 + r16) * 128) + kb * 64 + hi * 16;
            af.q[0] = *(const uint4*)rp;
            af.q[1] = *(const uint4*)(rp + 32);
        }
        Frag bfr[4];
#pragma unroll
        for (int ni = 0; ni < 4; ++ni) {
            const unsigned char* rp =
                (const unsigned char*)(Ub + (n0 + ni * 16 + r16) * 128) + kb * 64 + hi * 16;
            bfr[ni].q[0] = *(const uint4*)rp;
            bfr[ni].q[1] = *(const uint4*)(rp + 32);
        }
#pragma unroll
        for (int ni = 0; ni < 4; ++ni)
            oacc[ni] = __builtin_amdgcn_wmma_f32_16x16x32_bf16(
                false, af.v, false, bfr[ni].v, (short)0, oacc[ni], false, false);
    }

    // ---- relu + coalesced f32 stores ----
#pragma unroll
    for (int r = 0; r < 8; ++r) {
        const int m = m0 + hi * 8 + r;
        float* orow = out + (size_t)(ib + m) * DD;
#pragma unroll
        for (int ni = 0; ni < 4; ++ni) {
            const int c = n0 + ni * 16 + r16;
            const float v = oacc[ni][r];
            orow[c] = v > 0.0f ? v : 0.0f;
        }
    }
}

extern "C" void kernel_launch(void* const* d_in, const int* in_sizes, int n_in,
                              void* d_out, int out_size, void* d_ws, size_t ws_size,
                              hipStream_t stream) {
    (void)in_sizes; (void)n_in; (void)out_size; (void)ws_size;
    const float* x   = (const float*)d_in[0];
    const int*   adj = (const int*)d_in[1];
    const float* U   = (const float*)d_in[2];
    float*       out = (float*)d_out;
    unsigned short* xT = (unsigned short*)d_ws;   // [128][16384] bf16 = 4 MB

    xpose_kernel<<<dim3(NN / 128, DD / 32), 256, 0, stream>>>(x, xT);
    gcn_kernel<<<dim3(NN / BM), 256, 0, stream>>>(adj, xT, U, out);
}
